// RPN_54314156425503
// MI455X (gfx1250) — matrix operand order repaired and checked by reading, not compiled, not run on hardware
//
#include <hip/hip_runtime.h>
#include <hip/hip_bf16.h>

typedef __attribute__((ext_vector_type(16))) _Float16 v16h;
typedef __attribute__((ext_vector_type(8)))  float    v8f;
typedef __attribute__((ext_vector_type(2)))  _Float16 hpair;
typedef __attribute__((ext_vector_type(4)))  unsigned int u32x4;
typedef __attribute__((ext_vector_type(8)))  int i32x8;
typedef __attribute__((ext_vector_type(4)))  int i32x4;

#define CIN   1280
#define HID   256
#define HH    32
#define WW    32
#define HW    1024      // 32*32
#define NA    9
#define NBOX  9216      // HW * NA
#define NSORT 16384     // next pow2 >= NBOX for bitonic sort
#define NMS_SMEM (NBOX*16 + NBOX)   // float4 bbox[NBOX] + u8 sup[NBOX] = 156672 B
#define NSTEP 360       // 9 kernel positions * (1280/32) channel chunks

// ---------------------------------------------------------------------------
// Kernel 1: conv1 (3x3, 1280->256) as implicit GEMM, M=256 N=1024 K=9*1280.
// Block = 256 threads (8 waves), block tile 64(M) x 64(N); each wave owns a
// 16x32 tile (2 accumulators, A-fragment reused -> 2 v_wmma per K-step).
// Software pipeline with double-buffered LDS: global loads for step i+1 are
// issued to registers before the WMMAs of step i, converted fp32->f16 and
// stored to the other LDS buffer after the WMMAs, one barrier per step.
// LDS tiles are stored pre-swizzled in per-lane WMMA fragment layout, so a
// fragment is one contiguous 32 B LDS read per lane (ds_load_b128 x2).
//
// gfx1250 16-bit A-fragment (ISA 7.12.2): lanes 0-15: row M=lane,
// K={0..7,16..23}; lanes 16-31: row M=lane-16, K={8..15,24..31}.
// Stored (contiguous k) element (r,c) of a 64x32 tile -> lane=(r&15)|((c>>4)<<4),
// pos=c&15, mtile=r>>4.
// B-fragment: lane holds column N=lane&15, K=(lane>>4)*16+i.
// ---------------------------------------------------------------------------
__global__ __launch_bounds__(256)
void conv1_wmma(const float* __restrict__ feat, const float* __restrict__ w1,
                const float* __restrict__ b1, _Float16* __restrict__ hout)
{
    __shared__ alignas(32) _Float16 Afrag[2][4][32][16];  // 2 x 4 KB
    __shared__ alignas(32) _Float16 Bfrag[2][4][32][16];  // 2 x 4 KB

    const int tid  = threadIdx.x;
    const int lane = tid & 31;
    const int wave = tid >> 5;          // 0..7
    const int bm   = blockIdx.x >> 4;   // 0..3   (M block of 64)
    const int bn   = blockIdx.x & 15;   // 0..15  (N block of 64)
    const int m0   = bm * 64;
    const int n0   = bn * 64;
    const int mt   = wave >> 1;         // 0..3
    const int nth  = wave & 1;          // 0..1 -> n-tiles {2*nth, 2*nth+1}

    v8f acc[2] = {};
    float ra0[4], ra1[4], rb0[4], rb1[4];   // staged globals for next step

    // load phase: pure global->register, batches into one load clause
    auto loadRegs = [&](int kkL, int c0L) {
        const int ky = kkL / 3 - 1;
        const int kx = kkL % 3 - 1;
        #pragma unroll
        for (int it = 0; it < 4; ++it) {
            int q  = tid + (it << 8);
            int r  = q >> 4;            // 0..63
            int c2 = (q & 15) << 1;     // even K
            const float* wp = w1 + ((size_t)((m0 + r) * CIN + (c0L + c2)) * 9 + kkL);
            ra0[it] = wp[0]; ra1[it] = wp[9];
        }
        #pragma unroll
        for (int it = 0; it < 4; ++it) {
            int q   = tid + (it << 8);
            int nl  = q & 63;
            int kc0 = (q >> 6) << 1;
            int n = n0 + nl;
            int y = n >> 5, x = n & 31;
            int iy = y + ky, ix = x + kx;
            bool ok = (iy >= 0) && (iy < HH) && (ix >= 0) && (ix < WW);
            const float* fp = feat + ((c0L + kc0) * HW + iy * WW + ix);
            rb0[it] = ok ? fp[0]  : 0.f;
            rb1[it] = ok ? fp[HW] : 0.f;
        }
    };
    // store phase: v_cvt_pk_f16_f32 + ds_store_b32, pre-swizzled layout
    auto storeRegs = [&](int buf) {
        #pragma unroll
        for (int it = 0; it < 4; ++it) {
            int q   = tid + (it << 8);
            int r   = q >> 4;
            int c2  = (q & 15) << 1;
            int hi  = c2 >> 3;
            int ln  = (r & 15) | ((c2 >> 4) << 4);
            int pos = c2 & 15;
            hpair pv; pv.x = (_Float16)ra0[it]; pv.y = (_Float16)ra1[it];
            *(hpair*)&Afrag[buf][r >> 4][ln][pos] = pv;
        }
        #pragma unroll
        for (int it = 0; it < 4; ++it) {
            int q   = tid + (it << 8);
            int nl  = q & 63;
            int kc0 = (q >> 6) << 1;
            int ln  = (nl & 15) | ((kc0 >> 4) << 4);
            int pos = kc0 & 15;
            hpair pv; pv.x = (_Float16)rb0[it]; pv.y = (_Float16)rb1[it];
            *(hpair*)&Bfrag[buf][nl >> 4][ln][pos] = pv;
        }
    };

    // prologue: stage step 0 into buffer 0
    loadRegs(0, 0);
    storeRegs(0);
    __syncthreads();

    int kkc = 0, c0c = 32;              // coordinates of step+1
    for (int step = 0; step < NSTEP; ++step) {
        const int  buf  = step & 1;
        const bool more = (step + 1) < NSTEP;
        if (more) {
            loadRegs(kkc, c0c);                       // globals in flight...
            __builtin_prefetch(w1 + ((size_t)(m0 * CIN + c0c) * 9), 0, 0);
        }

        // ...while the MMAs run on the current buffer
        v16h a   = *(const v16h*)(&Afrag[buf][mt][lane][0]);
        v16h bvA = *(const v16h*)(&Bfrag[buf][nth * 2 + 0][lane][0]);
        v16h bvB = *(const v16h*)(&Bfrag[buf][nth * 2 + 1][lane][0]);
        acc[0] = __builtin_amdgcn_wmma_f32_16x16x32_f16(
            false, a, false, bvA, (short)0, acc[0], false, false);
        acc[1] = __builtin_amdgcn_wmma_f32_16x16x32_f16(
            false, a, false, bvB, (short)0, acc[1], false, false);

        if (more) storeRegs(buf ^ 1);
        c0c += 32;
        if (c0c == CIN) { c0c = 0; ++kkc; }
        __syncthreads();
    }

    // C/D layout: M = (lane>>4)*8 + r, N = lane&15
    const int mbase = m0 + mt * 16 + (lane >> 4) * 8;
    const int nnl   = lane & 15;
    #pragma unroll
    for (int j = 0; j < 2; ++j) {
        int nn = n0 + (nth * 2 + j) * 16 + nnl;
        #pragma unroll
        for (int r = 0; r < 8; ++r) {
            int m   = mbase + r;
            float v = acc[j][r] + b1[m];
            v = (v > 0.f) ? v : 0.01f * v;           // leaky_relu(0.01)
            hout[m * HW + nn] = (_Float16)v;
        }
    }
}

// ---------------------------------------------------------------------------
// Kernel 2: conv2 (1x1, 256->54, M padded to 64, K=256) + fused box decode.
// - w2 is staged ONCE into LDS in pre-swizzled f16 fragment layout (32 KB).
// - B-operand tiles (h, already f16) are DMA'd into LDS by the Tensor Data
//   Mover: 32x32 f16 tile, 64 B lines, line stride HW*2 B. Wave 0 issues
//   tensor_load_to_lds with a hand-packed D# and waits on TENSORcnt.
// - Each block holds all 54 output channels for its 32 pixels, so the
//   sigmoid/exp anchor decode runs straight out of LDS (no out2 buffer).
// ---------------------------------------------------------------------------
__global__ __launch_bounds__(256)
void conv2_wmma(const _Float16* __restrict__ h, const float* __restrict__ w2,
                const float* __restrict__ b2, const float* __restrict__ anchors,
                float* __restrict__ boxes, float* __restrict__ conf)
{
    __shared__ alignas(32) _Float16 Wf[8][4][32][16];  // w2 fragments, 32 KB
    __shared__ alignas(16) _Float16 Bh[32 * 32];       // TDM tile [k][n], 2 KB
    __shared__ float Sout[54][33];                     // block outputs, ~7 KB

    const int tid  = threadIdx.x;
    const int lane = tid & 31;
    const int wave = tid >> 5;
    const int mt   = wave >> 1;                 // 0..3 -> M tile
    const int nt   = wave & 1;
    const int n0   = blockIdx.x * 32;

    const int kbB  = (lane >> 4) * 16;
    const int ncol = nt * 16 + (lane & 15);     // 0..31 within tile

    // stage w2 once: 64(M,pad) x 256(K) -> f16 fragment layout
    #pragma unroll
    for (int q = tid; q < 8192; q += 256) {
        int r  = q >> 7;              // 0..63
        int c2 = (q & 127) << 1;      // even K, 0..254
        float v0 = 0.f, v1 = 0.f;
        if (r < 54) {
            float2 w = *(const float2*)&w2[r * HID + c2];
            v0 = w.x; v1 = w.y;
        }
        int kb  = c2 >> 5;
        int cc  = c2 & 31;
        int hi  = cc >> 3;
        int ln  = (r & 15) | ((cc >> 4) << 4);
        int pos = cc & 15;
        hpair pv; pv.x = (_Float16)v0; pv.y = (_Float16)v1;
        *(hpair*)&Wf[kb][r >> 4][ln][pos] = pv;
    }

    v8f acc = {};
    for (int k0 = 0; k0 < HID; k0 += 32) {
        __syncthreads();                        // prev tile consumed / Wf ready
        if (wave == 0) {
            // ---- Tensor DMA descriptor (D#) ----
            unsigned lds = (unsigned)(size_t)(&Bh[0]);
            unsigned long long ga =
                (unsigned long long)(size_t)(h + (size_t)k0 * HW + n0);
            u32x4 g0;
            g0[0] = 1u;                                   // count=1 (valid, load)
            g0[1] = lds;                                  // lds_addr
            g0[2] = (unsigned)ga;                         // global_addr[31:0]
            g0[3] = (unsigned)((ga >> 32) & 0x01FFFFFFu)  // global_addr[56:32]
                  | 0x80000000u;                          // type=2 ("image")
            i32x8 g1;
            g1[0] = (int)(1u << 16);       // workgroup_mask=0, data_size=1 (2 B)
            g1[1] = (int)((unsigned)HW << 16);   // tensor_dim0[15:0]=1024
            g1[2] = (int)((unsigned)HID << 16);  // dim0_hi=0 | tensor_dim1=256
            g1[3] = (int)(32u << 16);      // dim1_hi=0 | tile_dim0=32
            g1[4] = 32;                    // tile_dim1=32, tile_dim2=0
            g1[5] = HW;                    // tensor_dim0_stride=1024 elements
            g1[6] = 0;
            g1[7] = 0;
            i32x4 gz = {0, 0, 0, 0};
#if __clang_major__ >= 23
            i32x8 gz8 = {0, 0, 0, 0, 0, 0, 0, 0};
            __builtin_amdgcn_tensor_load_to_lds(g0, g1, gz, gz, gz8, 0);
#else
            __builtin_amdgcn_tensor_load_to_lds(g0, g1, gz, gz, 0);
#endif
            __builtin_amdgcn_s_wait_tensorcnt(0);
        }
        __syncthreads();                        // tile visible to all waves

        v16h a = *(const v16h*)(&Wf[k0 >> 5][mt][lane][0]);
        v16h b;
        #pragma unroll
        for (int i = 0; i < 16; ++i)
            b[i] = Bh[(kbB + i) * 32 + ncol];
        acc = __builtin_amdgcn_wmma_f32_16x16x32_f16(
            false, a, false, b, (short)0, acc, false, false);
    }

    // epilogue: all 54 channels for this block's 32 pixels -> LDS
    const int mbase = mt * 16 + (lane >> 4) * 8;
    #pragma unroll
    for (int r = 0; r < 8; ++r) {
        int m = mbase + r;
        if (m < 54) Sout[m][ncol] = acc[r] + b2[m];
    }
    __syncthreads();

    // fused anchor decode: 32 pixels x 9 anchors = 288 boxes per block
    for (int t = tid; t < 32 * NA; t += 256) {
        int nl = t / NA;
        int aa = t - nl * NA;
        int p  = n0 + nl;
        int y  = p >> 5, x = p & 31;

        float z  = Sout[aa * 6 + 0][nl];
        float cf = 1.f / (1.f + __expf(-z));
        float tx = Sout[aa * 6 + 2][nl];
        float ty = Sout[aa * 6 + 3][nl];
        float tw = Sout[aa * 6 + 4][nl];
        float th = Sout[aa * 6 + 5][nl];
        float aw = anchors[aa * 2 + 0], ah = anchors[aa * 2 + 1];

        float pcx = ((float)x + 0.5f) + tx * aw;
        float pcy = ((float)y + 0.5f) + ty * ah;
        float pw  = aw * __expf(tw);
        float ph  = ah * __expf(th);

        int i = p * NA + aa;                   // matches (y*32+x)*9 + a
        boxes[i * 4 + 0] = pcx - pw * 0.5f;
        boxes[i * 4 + 1] = pcy - ph * 0.5f;
        boxes[i * 4 + 2] = pcx + pw * 0.5f;
        boxes[i * 4 + 3] = pcy + ph * 0.5f;
        conf[i] = cf;
    }
}

// ---------------------------------------------------------------------------
// Kernel 3: NMS. Single workgroup, 1024 threads, big-LDS (CDNA5 320KB/WGP):
//   phase 1: bitonic sort of 16384 packed u64 keys (desc score | orig idx)
//   phase 2: gather sorted boxes into LDS, greedy sequential suppression
//   phase 3: scatter [box*k, conf*k] back to original order
// ---------------------------------------------------------------------------
__global__ __launch_bounds__(1024)
void nms_kernel(const float* __restrict__ boxes, const float* __restrict__ conf,
                int* __restrict__ sidx, float* __restrict__ out)
{
    extern __shared__ unsigned char smem[];
    unsigned long long* keys = (unsigned long long*)smem;
    const int tid = threadIdx.x;
    const int NT  = 1024;

    // build sort keys: ascending key order == descending score, stable by idx
    for (int i = tid; i < NSORT; i += NT) {
        unsigned long long key = 0xFFFFFFFFFFFFFFFFull;
        if (i < NBOX) {
            float c = conf[i];
            float s = (c > 0.5f) ? c : -1.0f;
            unsigned u    = __float_as_uint(s);
            unsigned asc  = (u & 0x80000000u) ? ~u : (u | 0x80000000u);
            unsigned desc = ~asc;
            key = ((unsigned long long)desc << 32) | (unsigned)i;
        }
        keys[i] = key;
    }
    __syncthreads();

    // bitonic sort (ascending)
    for (int k = 2; k <= NSORT; k <<= 1) {
        for (int j = k >> 1; j > 0; j >>= 1) {
            for (int t = tid; t < NSORT; t += NT) {
                int partner = t ^ j;
                if (partner > t) {
                    bool up = ((t & k) == 0);
                    unsigned long long av = keys[t], bv = keys[partner];
                    if ((av > bv) == up) { keys[t] = bv; keys[partner] = av; }
                }
            }
            __syncthreads();
        }
    }

    // spill sorted original indices to global (LDS gets reused next phase)
    for (int i = tid; i < NBOX; i += NT)
        sidx[i] = (int)(unsigned)(keys[i] & 0xFFFFFFFFu);
    __syncthreads();

    // re-carve LDS: sorted boxes + suppression flags
    float4*        bbox = (float4*)smem;
    unsigned char* sup  = (unsigned char*)(smem + NBOX * 16);
    for (int i = tid; i < NBOX; i += NT) {
        int orig = sidx[i];
        const float* bp = boxes + orig * 4;
        bbox[i] = make_float4(bp[0], bp[1], bp[2], bp[3]);
        sup[i]  = (conf[orig] > 0.5f) ? 0 : 1;     // invalid => pre-suppressed
    }
    __syncthreads();

    // greedy suppression, highest score first (matches reference fori_loop)
    for (int i = 0; i < NBOX; ++i) {
        if (!sup[i]) {                                  // uniform LDS read
            float4 bi = bbox[i];
            float  ai = fmaxf(bi.z - bi.x, 0.f) * fmaxf(bi.w - bi.y, 0.f);
            for (int j = i + 1 + tid; j < NBOX; j += NT) {
                if (!sup[j]) {
                    float4 bj = bbox[j];
                    float  aj = fmaxf(bj.z - bj.x, 0.f) * fmaxf(bj.w - bj.y, 0.f);
                    float iw = fminf(bi.z, bj.z) - fmaxf(bi.x, bj.x);
                    float ih = fminf(bi.w, bj.w) - fmaxf(bi.y, bj.y);
                    iw = fmaxf(iw, 0.f); ih = fmaxf(ih, 0.f);
                    float inter = iw * ih;
                    float iou   = inter / (ai + aj - inter + 1e-8f);
                    if (iou > 0.7f) sup[j] = 1;
                }
            }
        }
        __syncthreads();
    }

    // emit [boxes*k, conf*k] in original order
    for (int i = tid; i < NBOX; i += NT) {
        int   orig = sidx[i];
        float k    = sup[i] ? 0.f : 1.f;
        float4 b   = bbox[i];
        out[orig * 5 + 0] = b.x * k;
        out[orig * 5 + 1] = b.y * k;
        out[orig * 5 + 2] = b.z * k;
        out[orig * 5 + 3] = b.w * k;
        out[orig * 5 + 4] = conf[orig] * k;
    }
}

// ---------------------------------------------------------------------------
// Host launcher. Inputs: feat, anchors_wh, w1, b1, w2, b2 (all fp32).
// Workspace layout (bytes):
//   h    : f16[256*1024]      @ 0        (512 KB)
//   boxes: f32[9216*4]        @ 524288   (144 KB)
//   conf : f32[9216]          @ 671744   ( 36 KB)
//   sidx : i32[9216]          @ 708608   ( 36 KB)
// ---------------------------------------------------------------------------
extern "C" void kernel_launch(void* const* d_in, const int* in_sizes, int n_in,
                              void* d_out, int out_size, void* d_ws, size_t ws_size,
                              hipStream_t stream)
{
    const float* feat    = (const float*)d_in[0];
    const float* anchors = (const float*)d_in[1];
    const float* w1      = (const float*)d_in[2];
    const float* b1      = (const float*)d_in[3];
    const float* w2      = (const float*)d_in[4];
    const float* b2      = (const float*)d_in[5];
    float* out = (float*)d_out;

    char* ws = (char*)d_ws;
    _Float16* h    = (_Float16*)(ws);
    float*    boxes= (float*)(ws + 524288);
    float*    conf = (float*)(ws + 671744);
    int*      sidx = (int*)(ws + 708608);

    conv1_wmma<<<64, 256, 0, stream>>>(feat, w1, b1, h);
    conv2_wmma<<<32, 256, 0, stream>>>(h, w2, b2, anchors, boxes, conf);

    (void)hipFuncSetAttribute((const void*)nms_kernel,
                              hipFuncAttributeMaxDynamicSharedMemorySize, NMS_SMEM);
    nms_kernel<<<1, 1024, NMS_SMEM, stream>>>(boxes, conf, sidx, out);
}
